// MHA_rotary_29609504539188
// MI455X (gfx1250) — matrix operand order repaired and checked by reading, not compiled, not run on hardware
//
#include <hip/hip_runtime.h>
#include <cstdint>

#define DEVI __device__ __forceinline__

typedef __attribute__((ext_vector_type(16))) __bf16 v16bf;
typedef __attribute__((ext_vector_type(8)))  __bf16 v8bf;
typedef __attribute__((ext_vector_type(4)))  __bf16 v4bf;
typedef __attribute__((ext_vector_type(8)))  float  v8f;
typedef __attribute__((ext_vector_type(4)))  unsigned int v4u;
typedef __attribute__((ext_vector_type(8)))  int v8i;
typedef __attribute__((ext_vector_type(4)))  int v4i;

// Problem constants (fixed by the reference)
constexpr int kB = 32;
constexpr int kT = 512;
constexpr int kC = 1024;
constexpr int kH = 16;
constexpr int kD = 64;
constexpr int kM = kB * kT;          // 16384 rows
constexpr int kNATTN = kH * kD;      // 1024

// ---------------------------------------------------------------------------
// CDNA5 feature detection
// ---------------------------------------------------------------------------
#if defined(__has_builtin)
#if __has_builtin(__builtin_amdgcn_tensor_load_to_lds)
#define HAVE_TDM 1
#endif
#endif

#if defined(__has_builtin) && __has_builtin(__builtin_amdgcn_s_wait_tensorcnt)
#define WAIT_TENSORCNT0() __builtin_amdgcn_s_wait_tensorcnt(0)
#else
#define WAIT_TENSORCNT0() asm volatile("s_wait_tensorcnt 0x0" ::: "memory")
#endif

// ---------------------------------------------------------------------------
// WMMA helpers
// ---------------------------------------------------------------------------
DEVI v8f wmma_bf16(v16bf a, v16bf b, v8f c) {
    return __builtin_amdgcn_wmma_f32_16x16x32_bf16(
        /*neg_a=*/false, a, /*neg_b=*/false, b,
        /*c_mod=*/(short)0, c, /*reuse_a=*/false, /*reuse_b=*/false);
}

DEVI v8f zero8() { return (v8f){0.f,0.f,0.f,0.f,0.f,0.f,0.f,0.f}; }

// A-fragment (16x32, 16-bit): lane l holds row M=l&15; elements 0..7 = K 8g..8g+7,
// elements 8..15 = K 16+8g..16+8g+7  (g = l>>4).  base points at the row start.
DEVI v16bf ld_fragA(const __bf16* base, int g) {
    v8bf lo = *(const v8bf*)(base + 8 * g);
    v8bf hi = *(const v8bf*)(base + 16 + 8 * g);
    v16bf r;
#pragma unroll
    for (int i = 0; i < 8; i++) { r[i] = lo[i]; r[i + 8] = hi[i]; }
    return r;
}

// B-fragment (32x16, 16-bit): lane l holds col N=l&15; elements 0..7 = K 8g..8g+7,
// elements 8..15 = K 16+8g..16+8g+7  (g = l>>4).
DEVI v16bf ld_fragB(const __bf16* base, int g) {
    v8bf lo = *(const v8bf*)(base + 8 * g);
    v8bf hi = *(const v8bf*)(base + 16 + 8 * g);
    v16bf r;
#pragma unroll
    for (int i = 0; i < 8; i++) { r[i] = lo[i]; r[i + 8] = hi[i]; }
    return r;
}

// ds_swizzle xor-reductions within 16-lane halves (masks < 16 never cross bit4)
DEVI float swz_xor(float x, int imm_result) { return x; }  // (unused placeholder)
#define SWZ(x, m) __int_as_float(__builtin_amdgcn_ds_swizzle(__float_as_int(x), ((m) << 10) | 0x1f))
DEVI float red_max16(float x) {
    x = fmaxf(x, SWZ(x, 1)); x = fmaxf(x, SWZ(x, 2));
    x = fmaxf(x, SWZ(x, 4)); x = fmaxf(x, SWZ(x, 8));
    return x;
}
DEVI float red_sum16(float x) {
    x += SWZ(x, 1); x += SWZ(x, 2); x += SWZ(x, 4); x += SWZ(x, 8);
    return x;
}

// ---------------------------------------------------------------------------
// Tensor Data Mover: 2-D tile (rows of tile_d0 elems, tile_d1 rows) -> LDS,
// bf16 elements, row padding in LDS via D# pad fields.
// ---------------------------------------------------------------------------
#ifdef HAVE_TDM
DEVI void tdm_load_2d(uint32_t lds_off, uint64_t gaddr,
                      uint32_t tensor_d0, uint32_t tensor_d1,
                      uint32_t tile_d0, uint32_t tile_d1,
                      uint32_t stride0, uint32_t pad_interval, uint32_t pad_amount) {
    v4u g0;
    g0[0] = 1u;                                             // count=1, user mode
    g0[1] = lds_off;                                        // LDS byte address
    g0[2] = (uint32_t)(gaddr & 0xFFFFFFFFu);                // global_addr[31:0]
    g0[3] = (uint32_t)((gaddr >> 32) & 0x1FFFFFFu) | (2u << 30);  // [56:32] | type=2
    v8i g1;
    g1[0] = (int)((1u << 16) |                              // data_size = 2 bytes
                  (1u << 20) |                              // pad_enable
                  (pad_interval << 22) | (pad_amount << 25));
    g1[1] = (int)((tensor_d0 & 0xFFFFu) << 16);             // tensor_dim0[15:0] @ bit48
    g1[2] = (int)((tensor_d0 >> 16) | ((tensor_d1 & 0xFFFFu) << 16));
    g1[3] = (int)((tensor_d1 >> 16) | (tile_d0 << 16));     // tile_dim0 @ bit112
    g1[4] = (int)(tile_d1 & 0xFFFFu);                       // tile_dim1; tile_dim2=0
    g1[5] = (int)stride0;                                   // tensor_dim0_stride[31:0]
    g1[6] = 0;                                              // stride hi / dim1_stride lo
    g1[7] = 0;
    v4i gz = (v4i){0, 0, 0, 0};
#if __clang_major__ >= 23
    v8i gz8 = (v8i){0, 0, 0, 0, 0, 0, 0, 0};
    __builtin_amdgcn_tensor_load_to_lds(g0, g1, gz, gz, gz8, 0);
#else
    __builtin_amdgcn_tensor_load_to_lds(g0, g1, gz, gz, 0);
#endif
}
#endif

// ---------------------------------------------------------------------------
// fp32 -> bf16 conversion / gather kernels
// ---------------------------------------------------------------------------
__global__ __launch_bounds__(256) void conv_f32_bf16(const float* __restrict__ s,
                                                     __bf16* __restrict__ d, int n4) {
    int i = blockIdx.x * 256 + threadIdx.x;
    if (i < n4) {
        float4 f = ((const float4*)s)[i];
        v4bf o; o[0] = (__bf16)f.x; o[1] = (__bf16)f.y; o[2] = (__bf16)f.z; o[3] = (__bf16)f.w;
        ((v4bf*)d)[i] = o;
    }
}

__global__ __launch_bounds__(256) void gather_rows_bf16(const float* __restrict__ emb,
                                                        const int* __restrict__ idx,
                                                        __bf16* __restrict__ dst) {
    int row = blockIdx.x;                          // 0 .. kM-1
    size_t src = (size_t)idx[row] * kC;
    float4 f = ((const float4*)(emb + src))[threadIdx.x];   // 256 thr * float4 == 1024
    v4bf o; o[0] = (__bf16)f.x; o[1] = (__bf16)f.y; o[2] = (__bf16)f.z; o[3] = (__bf16)f.w;
    ((v4bf*)(dst + (size_t)row * kC))[threadIdx.x] = o;
}

// ---------------------------------------------------------------------------
// GEMM: C[16384,1024] = A(bf16) @ B(bf16) + bias.  128x128 block, 8 waves,
// each wave owns 32 rows x 64 cols (2 A-frags x 4 B-frags = 8 WMMA per K-step).
// MODE 0: out bf16 (B,H,T,D), no RoPE (V) | MODE 1: + RoPE (Q,K) | MODE 2: f32.
// A tile is DMA'd into LDS by the TDM (with hardware row padding) when available.
// ---------------------------------------------------------------------------
template <int MODE>
__global__ __launch_bounds__(256) void gemm_bf16_wmma(const __bf16* __restrict__ A,
                                                      const __bf16* __restrict__ Bw,
                                                      const float* __restrict__ bias,
                                                      __bf16* __restrict__ outB,
                                                      float* __restrict__ outF) {
    constexpr int LDSA = 40;   // 32 elems (64B) + 8 elems (16B) pad; rows 16B-aligned
    constexpr int LDSB = 40;
    __shared__ alignas(16) __bf16 As[128 * LDSA];   // 128 x 32 tile of A
    __shared__ alignas(16) __bf16 Bst[128 * LDSB];  // transposed 32 x 128 tile: Bst[n][k]

    const int tid = threadIdx.x;
    const int wave = tid >> 5, lane = tid & 31;
    const int g = lane >> 4, r = lane & 15;
    const int wr = (wave & 3) * 32;                 // wave row offset in block tile
    const int wc = (wave >> 2) * 64;                // wave col offset in block tile
    const int colBase = blockIdx.x * 128;
    const int rowBase = blockIdx.y * 128;
    const int K = kC, N = kC;

    v8f acc[2][4];
#pragma unroll
    for (int s = 0; s < 2; s++)
#pragma unroll
        for (int i = 0; i < 4; i++) acc[s][i] = zero8();

    const int blr = tid >> 4, blc = (tid & 15) * 8;   // B loader: 16 thr/row
#ifndef HAVE_TDM
    const int alr = tid >> 2, alc = (tid & 3) * 8;    // A loader: 4 thr/row
#endif

    for (int kb = 0; kb < K; kb += 32) {
#ifdef HAVE_TDM
        if (wave == 0)
            tdm_load_2d((uint32_t)(uintptr_t)As,
                        (uint64_t)(uintptr_t)(A + (size_t)rowBase * K + kb),
                        /*tensor_d0=*/K, /*tensor_d1=*/1u << 20,
                        /*tile_d0=*/32, /*tile_d1=*/128, /*stride0=*/K,
                        /*pad_interval: 16 dwords=*/3, /*pad_amount: 4 dwords=*/3);
#else
#pragma unroll
        for (int rr = alr; rr < 128; rr += 64)
            *(v8bf*)&As[rr * LDSA + alc] =
                *(const v8bf*)(A + (size_t)(rowBase + rr) * K + kb + alc);
#endif
#pragma unroll
        for (int rr = blr; rr < 32; rr += 16) {
            v8bf tv = *(const v8bf*)(Bw + (size_t)(kb + rr) * N + colBase + blc);
#pragma unroll
            for (int e = 0; e < 8; e++) Bst[(blc + e) * LDSB + rr] = tv[e];
        }
        if (kb + 32 < K)
            __builtin_prefetch(Bw + (size_t)(kb + 32 + blr) * N + colBase, 0, 0);
#ifdef HAVE_TDM
        WAIT_TENSORCNT0();
#endif
        __syncthreads();

        v16bf a0 = ld_fragA(&As[(wr + r) * LDSA], g);
        v16bf a1 = ld_fragA(&As[(wr + 16 + r) * LDSA], g);
#pragma unroll
        for (int ct = 0; ct < 4; ct++) {
            v16bf bfrag = ld_fragB(&Bst[(wc + ct * 16 + r) * LDSB], g);
            acc[0][ct] = wmma_bf16(a0, bfrag, acc[0][ct]);
            acc[1][ct] = wmma_bf16(a1, bfrag, acc[1][ct]);
        }
        __syncthreads();
    }

    // ---- epilogue -----------------------------------------------------------
    float bv[4];
#pragma unroll
    for (int ct = 0; ct < 4; ct++) bv[ct] = bias[colBase + wc + ct * 16 + r];

#pragma unroll
    for (int sub = 0; sub < 2; sub++) {
        const int rowW = rowBase + wr + sub * 16;
#pragma unroll
        for (int ct = 0; ct < 4; ct++)
#pragma unroll
            for (int j = 0; j < 8; j++) acc[sub][ct][j] += bv[ct];

        if constexpr (MODE == 2) {
#pragma unroll
            for (int ct = 0; ct < 4; ct++) {
                int col = colBase + wc + ct * 16 + r;
#pragma unroll
                for (int j = 0; j < 8; j++) {
                    int row = rowW + j + 8 * g;
                    outF[(size_t)row * N + col] = acc[sub][ct][j];
                }
            }
        } else {
            if constexpr (MODE == 1) {
                // RoPE on head dims 0..31 (tiles ct=0,1): pair (d, d+16), freq idx = r
                float invf = __powf(10000.0f, -(float)r * (1.0f / 16.0f));
#pragma unroll
                for (int j = 0; j < 8; j++) {
                    int row = rowW + j + 8 * g;
                    int t = row & (kT - 1);
                    float ang = (float)t * invf;
                    float cs = __cosf(ang), sn = __sinf(ang);
                    float a0 = acc[sub][0][j], a1 = acc[sub][1][j];
                    acc[sub][0][j] = a0 * cs - a1 * sn;
                    acc[sub][1][j] = a1 * cs + a0 * sn;
                }
            }
            int h = (colBase + wc) >> 6;            // 64-wide span == one head
#pragma unroll
            for (int ct = 0; ct < 4; ct++) {
                int d = ct * 16 + r;
#pragma unroll
                for (int j = 0; j < 8; j++) {
                    int row = rowW + j + 8 * g;
                    int b = row >> 9;
                    int t = row & (kT - 1);
                    outB[(((size_t)b * kH + h) * kT + t) * kD + d] = (__bf16)acc[sub][ct][j];
                }
            }
        }
    }
}

// ---------------------------------------------------------------------------
// Flash attention: grid = B*H*(T/64) blocks, 128 threads (4 waves).
// Wave w owns 16 query rows; workgroup shares 32-key K/V blocks through LDS.
// K block is TDM-DMA'd (hardware padding); V is transposed by the loaders.
// ---------------------------------------------------------------------------
__global__ __launch_bounds__(128) void attn_wmma(const __bf16* __restrict__ qp,
                                                 const __bf16* __restrict__ kp,
                                                 const __bf16* __restrict__ vp,
                                                 __bf16* __restrict__ op) {
    constexpr int LDK = 72;     // 64 elems (128B) + 8 elems (16B) pad
    constexpr int LDV = 40;
    constexpr int LDP = 40;
    __shared__ alignas(16) __bf16 Ks[32 * LDK];       // K block: [key][d]
    __shared__ alignas(16) __bf16 VTs[64 * LDV];      // V block transposed: [d][key]
    __shared__ alignas(16) __bf16 Ps[4 * 16 * LDP];   // per-wave P staging

    const int tid = threadIdx.x;
    const int wave = tid >> 5, lane = tid & 31;
    const int g = lane >> 4, r = lane & 15;
    const int bh = blockIdx.x >> 3;                   // b*H + h
    const int qt = blockIdx.x & 7;                    // 64-row query block within T
    const int q0 = qt * 64 + wave * 16;               // this wave's first query row

    const __bf16* qbase = qp + ((size_t)bh * kT + q0 + r) * kD;
    v16bf qa0 = ld_fragA(qbase, g);
    v16bf qa1 = ld_fragA(qbase + 32, g);

    v8f oacc[4];
#pragma unroll
    for (int i = 0; i < 4; i++) oacc[i] = zero8();
    float mrow[8], lrow[8];
#pragma unroll
    for (int j = 0; j < 8; j++) { mrow[j] = -1e30f; lrow[j] = 0.0f; }

    __bf16* Pw = &Ps[wave * 16 * LDP];
    const int ldr = tid >> 3, ldc = (tid & 7) * 8;    // 8 thr/row cooperative loaders
    const int nkb = qt * 2 + 2;                       // causal key-block count

    for (int kb = 0; kb < nkb; ++kb) {
        const int k0 = kb * 32;
#ifdef HAVE_TDM
        if (wave == 0)
            tdm_load_2d((uint32_t)(uintptr_t)Ks,
                        (uint64_t)(uintptr_t)(kp + ((size_t)bh * kT + k0) * kD),
                        /*tensor_d0=*/kD, /*tensor_d1=*/1u << 20,
                        /*tile_d0=*/64, /*tile_d1=*/32, /*stride0=*/kD,
                        /*pad_interval: 32 dwords=*/4, /*pad_amount: 4 dwords=*/3);
#endif
#pragma unroll
        for (int rr = ldr; rr < 32; rr += 16) {
#ifndef HAVE_TDM
            *(v8bf*)&Ks[rr * LDK + ldc] =
                *(const v8bf*)(kp + ((size_t)bh * kT + k0 + rr) * kD + ldc);
#endif
            v8bf tv = *(const v8bf*)(vp + ((size_t)bh * kT + k0 + rr) * kD + ldc);
#pragma unroll
            for (int e = 0; e < 8; e++) VTs[(ldc + e) * LDV + rr] = tv[e];
        }
#ifdef HAVE_TDM
        WAIT_TENSORCNT0();
#endif
        __syncthreads();

        if (k0 <= q0) {   // wave-uniform: EXEC stays all-ones around WMMA
            v8f s0 = zero8(), s1 = zero8();
            s0 = wmma_bf16(qa0, ld_fragB(&Ks[r * LDK], g), s0);
            s0 = wmma_bf16(qa1, ld_fragB(&Ks[r * LDK + 32], g), s0);
            s1 = wmma_bf16(qa0, ld_fragB(&Ks[(16 + r) * LDK], g), s1);
            s1 = wmma_bf16(qa1, ld_fragB(&Ks[(16 + r) * LDK + 32], g), s1);

#pragma unroll
            for (int j = 0; j < 8; j++) {
                int qg = q0 + j + 8 * g;
                float x0 = s0[j] * 0.125f;            // 1/sqrt(64)
                float x1 = s1[j] * 0.125f;
                if (k0 + r > qg)       x0 = -1e30f;
                if (k0 + 16 + r > qg)  x1 = -1e30f;
                float rmax = red_max16(fmaxf(x0, x1));
                float mnew = fmaxf(mrow[j], rmax);
                float fac = __expf(mrow[j] - mnew);
                mrow[j] = mnew;
                float p0 = __expf(x0 - mnew);
                float p1 = __expf(x1 - mnew);
                float rs = red_sum16(p0 + p1);
                lrow[j] = lrow[j] * fac + rs;
#pragma unroll
                for (int ct = 0; ct < 4; ct++) oacc[ct][j] *= fac;
                int m = j + 8 * g;
                Pw[m * LDP + r] = (__bf16)p0;
                Pw[m * LDP + 16 + r] = (__bf16)p1;
            }
            v16bf pf = ld_fragA(&Pw[r * LDP], g);
#pragma unroll
            for (int ct = 0; ct < 4; ct++) {
                v16bf vf = ld_fragB(&VTs[(ct * 16 + r) * LDV], g);
                oacc[ct] = wmma_bf16(pf, vf, oacc[ct]);
            }
        }
        __syncthreads();
    }

    const int b = bh >> 4, h = bh & 15;
#pragma unroll
    for (int j = 0; j < 8; j++) {
        float inv = 1.0f / lrow[j];
        int t = q0 + j + 8 * g;
        size_t rowoff = ((size_t)b * kT + t) * kNATTN + h * kD;
#pragma unroll
        for (int ct = 0; ct < 4; ct++)
            op[rowoff + ct * 16 + r] = (__bf16)(oacc[ct][j] * inv);
    }
}

// ---------------------------------------------------------------------------
// Final elementwise: pos_logits = lf * emb[pos]; neg_logits = sum(lf * emb[neg])
// ---------------------------------------------------------------------------
__global__ __launch_bounds__(256) void final_logits(const float* __restrict__ lf,
                                                    const float* __restrict__ emb,
                                                    const int* __restrict__ pos,
                                                    const int* __restrict__ neg,
                                                    float* __restrict__ outPos,
                                                    float* __restrict__ outNeg) {
    __shared__ float red[8];
    const int row = blockIdx.x, tid = threadIdx.x;
    float4 a = ((const float4*)(lf + (size_t)row * kC))[tid];
    float4 p = ((const float4*)(emb + (size_t)pos[row] * kC))[tid];
    float4 n = ((const float4*)(emb + (size_t)neg[row] * kC))[tid];
    float4 pl;
    pl.x = a.x * p.x; pl.y = a.y * p.y; pl.z = a.z * p.z; pl.w = a.w * p.w;
    ((float4*)(outPos + (size_t)row * kC))[tid] = pl;
    float s = a.x * n.x + a.y * n.y + a.z * n.z + a.w * n.w;
#pragma unroll
    for (int off = 16; off; off >>= 1) s += __shfl_xor(s, off, 32);
    if ((tid & 31) == 0) red[tid >> 5] = s;
    __syncthreads();
    if (tid == 0) {
        float t = 0.f;
#pragma unroll
        for (int i = 0; i < 8; i++) t += red[i];
        outNeg[row] = t;
    }
}

// ---------------------------------------------------------------------------
// Launch
// ---------------------------------------------------------------------------
extern "C" void kernel_launch(void* const* d_in, const int* in_sizes, int n_in,
                              void* d_out, int out_size, void* d_ws, size_t ws_size,
                              hipStream_t stream) {
    (void)in_sizes; (void)n_in; (void)out_size; (void)ws_size;
    const int*   log_seqs = (const int*)d_in[1];
    const int*   pos_seqs = (const int*)d_in[2];
    const int*   neg_seqs = (const int*)d_in[3];
    const float* item_emb = (const float*)d_in[4];
    const float* Wq = (const float*)d_in[5];  const float* bq = (const float*)d_in[6];
    const float* Wk = (const float*)d_in[7];  const float* bk = (const float*)d_in[8];
    const float* Wv = (const float*)d_in[9];  const float* bv = (const float*)d_in[10];
    const float* Wo = (const float*)d_in[11]; const float* bo = (const float*)d_in[12];
    float* out = (float*)d_out;

    // workspace layout
    char* base = (char*)d_ws;
    size_t off = 0;
    auto alloc_bf = [&](size_t elems) { __bf16* p = (__bf16*)(base + off); off += elems * 2; return p; };
    const size_t WSZ = (size_t)kC * kC;        // 1M
    const size_t XSZ = (size_t)kM * kC;        // 16M
    __bf16* wq_b = alloc_bf(WSZ);
    __bf16* wk_b = alloc_bf(WSZ);
    __bf16* wv_b = alloc_bf(WSZ);
    __bf16* wo_b = alloc_bf(WSZ);
    __bf16* x_b  = alloc_bf(XSZ);
    __bf16* q_b  = alloc_bf(XSZ);              // (B,H,T,D)
    __bf16* k_b  = alloc_bf(XSZ);
    __bf16* v_b  = alloc_bf(XSZ);
    __bf16* o_b  = alloc_bf(XSZ);              // (B,T,H*D)
    float*  lf   = (float*)(base + off); off += XSZ * 4;

    // 1) weight conversion + embedding gather
    const int n4w = (int)(WSZ / 4);
    conv_f32_bf16<<<(n4w + 255) / 256, 256, 0, stream>>>(Wq, wq_b, n4w);
    conv_f32_bf16<<<(n4w + 255) / 256, 256, 0, stream>>>(Wk, wk_b, n4w);
    conv_f32_bf16<<<(n4w + 255) / 256, 256, 0, stream>>>(Wv, wv_b, n4w);
    conv_f32_bf16<<<(n4w + 255) / 256, 256, 0, stream>>>(Wo, wo_b, n4w);
    gather_rows_bf16<<<kM, 256, 0, stream>>>(item_emb, log_seqs, x_b);

    // 2) QKV projections (RoPE fused into Q/K epilogues)
    dim3 ggrid(kC / 128, kM / 128);
    gemm_bf16_wmma<1><<<ggrid, 256, 0, stream>>>(x_b, wq_b, bq, q_b, nullptr);
    gemm_bf16_wmma<1><<<ggrid, 256, 0, stream>>>(x_b, wk_b, bk, k_b, nullptr);
    gemm_bf16_wmma<0><<<ggrid, 256, 0, stream>>>(x_b, wv_b, bv, v_b, nullptr);

    // 3) causal flash attention
    attn_wmma<<<kB * kH * (kT / 64), 128, 0, stream>>>(q_b, k_b, v_b, o_b);

    // 4) output projection -> f32 log_feats
    gemm_bf16_wmma<2><<<ggrid, 256, 0, stream>>>(o_b, wo_b, bo, nullptr, lf);

    // 5) pos/neg logits
    final_logits<<<kM, 256, 0, stream>>>(lf, item_emb, pos_seqs, neg_seqs,
                                         out, out + (size_t)kM * kC);
}
